// TacticalGNN_45311904973201
// MI455X (gfx1250) — matrix operand order repaired and checked by reading, not compiled, not run on hardware
//
#include <hip/hip_runtime.h>
#include <hip/hip_bf16.h>
#include <math.h>

#define Nn 50000
#define En 800000
#define Gn 2048
#define DIN 12
#define DH 128
#define DOUT 64
#define DEDGE 5

typedef __attribute__((ext_vector_type(16))) _Float16 v16h;
typedef __attribute__((ext_vector_type(8)))  float    v8f;

// ---------------------------------------------------------------- utilities
__device__ __forceinline__ void atomicMaxF32(float* addr, float val) {
    unsigned int* ua = (unsigned int*)addr;
    unsigned int old = *ua;
    while (__uint_as_float(old) < val) {
        unsigned int assumed = old;
        old = atomicCAS(ua, assumed, __float_as_uint(val));
        if (old == assumed) break;
    }
}

__global__ void fill_f32_k(float* p, float v, int n) {
    int t = blockIdx.x * blockDim.x + threadIdx.x;
    if (t < n) p[t] = v;
}
__global__ void fill_i32_k(int* p, int v, int n) {
    int t = blockIdx.x * blockDim.x + threadIdx.x;
    if (t < n) p[t] = v;
}
// out[r, :] = bias[:]  (pre-loads aggregation bias so scatter-add lands on it)
__global__ void init_rows_bias_k(float* out, const float* __restrict__ bias, int rows, int stride) {
    int t = blockIdx.x * blockDim.x + threadIdx.x;
    if (t < rows * stride) out[t] = bias[t % stride];
}

// Transpose + f32->f16 convert a weight matrix: W[K,Nout] -> Wt[Nout,K] (f16).
// Tiny (<=16K elems), stays hot in L2 for the following GEMM.
__global__ void wconv_k(const float* __restrict__ W, _Float16* __restrict__ Wt, int K, int Nout) {
    int t = blockIdx.x * blockDim.x + threadIdx.x;
    if (t >= K * Nout) return;
    int k = t / Nout, n = t - k * Nout;
    Wt[n * K + k] = (_Float16)W[t];
}

// ------------------------------------------------- WMMA f16->f32 GEMM
// C[M,Nout] = act(A[M,K] @ W[K,Nout] + bias). One wave32 per 16x16 tile.
// Requires M % 16 == 0 (true for all call sites: 50000, 2048).
// A fragment (ISA 7.12.2, 16x32 f16): per lane two contiguous 8-float runs ->
// 4x b128 loads + pack-convert. B fragment: Wt is pre-transposed f16, so the
// 16 elements (k = kb + hi*8 + 0..7, kb + 16 + hi*8 + 0..7) are two contiguous v8h runs.
template <int K>
__global__ void gemm_wmma_f16_k(const float* __restrict__ A, const _Float16* __restrict__ Wt,
                                const float* __restrict__ bias, float* __restrict__ C,
                                int M, int act, int tilesN) {
    int wave = blockIdx.x * (blockDim.x >> 5) + (threadIdx.x >> 5);
    int tilesM = M >> 4;
    if (wave >= tilesM * tilesN) return;
    int tm = wave / tilesN, tn = wave - tm * tilesN;
    int lane = threadIdx.x & 31;
    int lo = lane & 15, hi = lane >> 4;
    int m = tm * 16 + lo;            // A row for this lane
    int n = tn * 16 + lo;            // B/C column for this lane
    const float*    Ap = A + (size_t)m * K;
    const _Float16* Bp = Wt + (size_t)n * K;
    v8f acc = {};
#pragma unroll
    for (int kb = 0; kb < K; kb += 32) {
        const float4* r0 = (const float4*)(Ap + kb + (hi << 3));        // k = kb+hi*8 .. +7
        const float4* r1 = (const float4*)(Ap + kb + 16 + (hi << 3));   // k = kb+16+hi*8 .. +7
        float4 a0 = r0[0], a1 = r0[1], a2 = r1[0], a3 = r1[1];
        v16h a;
        a[0]  = (_Float16)a0.x; a[1]  = (_Float16)a0.y; a[2]  = (_Float16)a0.z; a[3]  = (_Float16)a0.w;
        a[4]  = (_Float16)a1.x; a[5]  = (_Float16)a1.y; a[6]  = (_Float16)a1.z; a[7]  = (_Float16)a1.w;
        a[8]  = (_Float16)a2.x; a[9]  = (_Float16)a2.y; a[10] = (_Float16)a2.z; a[11] = (_Float16)a2.w;
        a[12] = (_Float16)a3.x; a[13] = (_Float16)a3.y; a[14] = (_Float16)a3.z; a[15] = (_Float16)a3.w;
        typedef __attribute__((ext_vector_type(8))) _Float16 v8h;
        v8h b0 = *(const v8h*)(Bp + kb + (hi << 3)), b1 = *(const v8h*)(Bp + kb + 16 + (hi << 3));
        v16h b = __builtin_shufflevector(b0, b1, 0, 1, 2, 3, 4, 5, 6, 7, 8, 9, 10, 11, 12, 13, 14, 15); // k = kb+hi*8 .. +7, kb+16+hi*8 .. +7
        acc = __builtin_amdgcn_wmma_f32_16x16x32_f16(
            /*neg_a=*/false, a, /*neg_b=*/false, b,
            /*c_mod=*/(short)0, acc, /*reuse_a=*/false, /*reuse_b=*/false);
    }
    int Nout = tilesN << 4;
    float bv = bias ? bias[n] : 0.f;
#pragma unroll
    for (int r = 0; r < 8; ++r) {
        int row = tm * 16 + r + (hi << 3);    // C/D: VGPR r -> row r (+8 for lane-hi)
        float v = acc[r] + bv;
        if (act) v = v > 0.f ? v : 0.f;
        C[(size_t)row * Nout + n] = v;
    }
}

// --------------------------------------------- input linear (K=12, scalar)
__global__ void lin_in_relu_k(const float* __restrict__ x, const float* __restrict__ W,
                              const float* __restrict__ b, float* __restrict__ h) {
    int t = blockIdx.x * blockDim.x + threadIdx.x;
    if (t >= Nn * DH) return;
    int nrow = t >> 7, j = t & 127;
    float s = b[j];
#pragma unroll
    for (int k = 0; k < DIN; ++k) s += x[nrow * DIN + k] * W[k * DH + j];
    h[t] = s > 0.f ? s : 0.f;
}

// --------------------------------------------- GATv2 edge pass 1: logits + segment max
// One wave per edge; fuses the 5xHC edge-attr projection; per-head dot with att;
// wave32 xor-shuffle reduction; atomicMax into per-(dst,head) running max.
__global__ void edge_logits_k(const float* __restrict__ xl, const float* __restrict__ xr,
                              const int* __restrict__ ei, const float* __restrict__ eattr,
                              const float* __restrict__ We, const float* __restrict__ be,
                              const float* __restrict__ att, float* __restrict__ logits,
                              float* __restrict__ mx, int H, int HC, int cshift) {
    int e = blockIdx.x * (blockDim.x >> 5) + (threadIdx.x >> 5);
    if (e >= En) return;
    int lane = threadIdx.x & 31;
    int src = ei[e], dst = ei[En + e];
    float ea0 = eattr[e * 5 + 0], ea1 = eattr[e * 5 + 1], ea2 = eattr[e * 5 + 2];
    float ea3 = eattr[e * 5 + 3], ea4 = eattr[e * 5 + 4];
    float part[4] = {0.f, 0.f, 0.f, 0.f};
    for (int ch = lane; ch < HC; ch += 32) {
        float ee = be[ch] + ea0 * We[ch] + ea1 * We[HC + ch] + ea2 * We[2 * HC + ch]
                          + ea3 * We[3 * HC + ch] + ea4 * We[4 * HC + ch];
        float v = xl[src * HC + ch] + xr[dst * HC + ch] + ee;
        v = v > 0.f ? v : 0.2f * v;                 // leaky_relu(0.2)
        part[ch >> cshift] += att[ch] * v;          // att is [H,C] flat
    }
#pragma unroll
    for (int off = 16; off > 0; off >>= 1) {
#pragma unroll
        for (int hh = 0; hh < 4; ++hh) part[hh] += __shfl_xor(part[hh], off, 32);
    }
    if (lane < H) {
        logits[e * H + lane] = part[lane];
        atomicMaxF32(&mx[dst * H + lane], part[lane]);
    }
}

// --------------------------------------------- edge pass 2: exp + segment sum
__global__ void edge_softmax_k(const float* __restrict__ logits, const float* __restrict__ mx,
                               const int* __restrict__ ei, float* __restrict__ ex,
                               float* __restrict__ denom, int H) {
    int t = blockIdx.x * blockDim.x + threadIdx.x;
    if (t >= En * H) return;
    int e = t / H, h = t - e * H;
    int dst = ei[En + e];
    float v = __expf(logits[t] - mx[dst * H + h]);
    ex[t] = v;
    atomicAdd(&denom[dst * H + h], v);
}

// --------------------------------------------- edge pass 3: weighted scatter-add
__global__ void edge_scatter_k(const float* __restrict__ xl, const int* __restrict__ ei,
                               const float* __restrict__ ex, const float* __restrict__ denom,
                               float* __restrict__ out, int H, int HC, int cshift) {
    int e = blockIdx.x;
    int ch = threadIdx.x;  // blockDim = HC
    int src = ei[e], dst = ei[En + e];
    int h = ch >> cshift;
    float alpha = ex[e * H + h] / denom[dst * H + h];
    atomicAdd(&out[dst * HC + ch], alpha * xl[src * HC + ch]);
}

// --------------------------------------------- batchnorm (batch stats) + relu + residual
__global__ void bn_stats_k(const float* __restrict__ h, float* __restrict__ colsum,
                           float* __restrict__ colsq) {
    int col = threadIdx.x;              // 128 threads
    int r0 = blockIdx.x * 128;
    float s = 0.f, sq = 0.f;
    int rend = r0 + 128 < Nn ? r0 + 128 : Nn;
    for (int r = r0; r < rend; ++r) {
        float v = h[r * DH + col];
        s += v; sq += v * v;
    }
    atomicAdd(&colsum[col], s);
    atomicAdd(&colsq[col], sq);
}

__global__ void bn_apply_k(float* __restrict__ hout, const float* __restrict__ hin,
                           const float* __restrict__ colsum, const float* __restrict__ colsq,
                           const float* __restrict__ g, const float* __restrict__ b) {
    int t = blockIdx.x * blockDim.x + threadIdx.x;
    if (t >= Nn * DH) return;
    int j = t & 127;
    float mean = colsum[j] * (1.f / Nn);
    float var  = colsq[j] * (1.f / Nn) - mean * mean;
    float v = (hout[t] - mean) * rsqrtf(var + 1e-5f) * g[j] + b[j];
    v = v > 0.f ? v : 0.f;
    hout[t] = v + hin[t];               // residual
}

// --------------------------------------------- graph pooling
__global__ void pool_accum_k(const float* __restrict__ ne, const int* __restrict__ batch,
                             float* __restrict__ gsum, float* __restrict__ gmax,
                             int* __restrict__ counts) {
    int t = blockIdx.x * blockDim.x + threadIdx.x;
    if (t >= Nn * DOUT) return;
    int nrow = t >> 6, j = t & 63;
    int g = batch[nrow];
    float v = ne[t];
    atomicAdd(&gsum[g * DOUT + j], v);
    atomicMaxF32(&gmax[g * DOUT + j], v);
    if (j == 0) atomicAdd(&counts[g], 1);
}

__global__ void pool_concat_k(const float* __restrict__ gsum, const float* __restrict__ gmax,
                              const int* __restrict__ counts, float* __restrict__ cc) {
    int t = blockIdx.x * blockDim.x + threadIdx.x;
    if (t >= Gn * 2 * DOUT) return;
    int g = t >> 7, j = t & 127;
    float cnt = (float)counts[g];
    float v;
    if (j < DOUT) v = gsum[g * DOUT + j] / fmaxf(cnt, 1.f);
    else { v = gmax[g * DOUT + (j - DOUT)]; v = cnt > 0.f ? v : 0.f; }
    cc[t] = v;
}

// ============================================================ host driver
extern "C" void kernel_launch(void* const* d_in, const int* in_sizes, int n_in,
                              void* d_out, int out_size, void* d_ws, size_t ws_size,
                              hipStream_t stream) {
    const float* x     = (const float*)d_in[0];
    const int*   ei    = (const int*)d_in[1];
    const float* eattr = (const float*)d_in[2];
    const int*   batch = (const int*)d_in[3];
    const float* W_in  = (const float*)d_in[4];
    const float* b_in  = (const float*)d_in[5];
    // gat_layers: 4 layers x (Wl, bl, Wr, br, We, be, att, bo) -> indices 6..37
    // bn_params: 3 x (g, b) -> 38..43 ; node_proj 44..47 ; graph_proj 48..51
    auto in = [&](int i) { return (const float*)d_in[i]; };

    float* ws = (float*)d_ws;
    size_t o = 0;
    auto alloc = [&](size_t nflt) { float* p = ws + o; o += nflt; return p; };
    float* bufA   = alloc((size_t)Nn * DH);
    float* bufB   = alloc((size_t)Nn * DH);
    float* xl     = alloc((size_t)Nn * DH);
    float* xr     = alloc((size_t)Nn * DH);
    float* logits = alloc((size_t)En * 4);
    float* exb    = alloc((size_t)En * 4);
    float* mx     = alloc((size_t)Nn * 4);
    float* denom  = alloc((size_t)Nn * 4);
    float* colsum = alloc(DH);
    float* colsq  = alloc(DH);
    float* gsum   = alloc((size_t)Gn * DOUT);
    float* gmaxb  = alloc((size_t)Gn * DOUT);
    int*   counts = (int*)alloc(Gn);
    float* cc     = alloc((size_t)Gn * 2 * DOUT);
    float* ghid   = alloc((size_t)Gn * DH);
    _Float16* Wt  = (_Float16*)alloc(DH * DH / 2 + 16);   // f16 transposed weights (<=32KB)
    float* ntmp   = xl;  // reuse for node-proj hidden [N,64]

    // GEMM: prep (transpose+f16 convert W) then WMMA kernel. M must be %16==0.
    auto gemm = [&](const float* A, const float* W, const float* bias, float* C,
                    int M, int K, int Nout, int act) {
        wconv_k<<<(K * Nout + 255) / 256, 256, 0, stream>>>(W, Wt, K, Nout);
        int tilesN = Nout / 16;
        int tiles = (M / 16) * tilesN;
        int blocks = (tiles + 3) / 4;             // 4 waves per 128-thread block
        if (K == 128)
            gemm_wmma_f16_k<128><<<blocks, 128, 0, stream>>>(A, Wt, bias, C, M, act, tilesN);
        else
            gemm_wmma_f16_k<64><<<blocks, 128, 0, stream>>>(A, Wt, bias, C, M, act, tilesN);
    };
    auto fillf = [&](float* p, float v, int n) {
        fill_f32_k<<<(n + 255) / 256, 256, 0, stream>>>(p, v, n);
    };

    // ---- input MLP: h = relu(x @ W_in + b_in)
    float* h = bufA; float* nxt = bufB;
    lin_in_relu_k<<<(Nn * DH + 255) / 256, 256, 0, stream>>>(x, W_in, b_in, h);

    // ---- 4 GATv2 layers
    for (int l = 0; l < 4; ++l) {
        int concat = (l < 3);
        int H  = concat ? 4 : 1;
        int HC = concat ? 128 : 64;
        int cshift = concat ? 5 : 6;
        const float* Wl  = in(6 + l * 8 + 0); const float* bl  = in(6 + l * 8 + 1);
        const float* Wr  = in(6 + l * 8 + 2); const float* br  = in(6 + l * 8 + 3);
        const float* We  = in(6 + l * 8 + 4); const float* be  = in(6 + l * 8 + 5);
        const float* att = in(6 + l * 8 + 6); const float* bo  = in(6 + l * 8 + 7);

        gemm(h, Wl, bl, xl, Nn, DH, HC, 0);                 // xl = h@Wl + bl
        gemm(h, Wr, br, xr, Nn, DH, HC, 0);                 // xr = h@Wr + br

        fillf(mx, -INFINITY, Nn * H);
        fillf(denom, 0.f, Nn * H);
        init_rows_bias_k<<<(Nn * HC + 255) / 256, 256, 0, stream>>>(nxt, bo, Nn, HC);

        edge_logits_k<<<(En + 7) / 8, 256, 0, stream>>>(xl, xr, ei, eattr, We, be, att,
                                                        logits, mx, H, HC, cshift);
        edge_softmax_k<<<(En * H + 255) / 256, 256, 0, stream>>>(logits, mx, ei, exb, denom, H);
        edge_scatter_k<<<En, HC, 0, stream>>>(xl, ei, exb, denom, nxt, H, HC, cshift);

        if (concat) {
            fillf(colsum, 0.f, DH);
            fillf(colsq, 0.f, DH);
            bn_stats_k<<<(Nn + 127) / 128, 128, 0, stream>>>(nxt, colsum, colsq);
            bn_apply_k<<<(Nn * DH + 255) / 256, 256, 0, stream>>>(
                nxt, h, colsum, colsq, in(38 + l * 2), in(38 + l * 2 + 1));
        }
        float* tsw = h; h = nxt; nxt = tsw;                 // h <- layer output
    }

    // ---- node projection (h now [N,64]): node_emb = relu(h@Wn1+bn1)@Wn2+bn2
    float* node_emb = (float*)d_out;                        // first N*64 of output
    gemm(h, in(44), in(45), ntmp, Nn, DOUT, DOUT, 1);
    gemm(ntmp, in(46), in(47), node_emb, Nn, DOUT, DOUT, 0);

    // ---- graph pooling (mean + max per graph)
    fillf(gsum, 0.f, Gn * DOUT);
    fillf(gmaxb, -INFINITY, Gn * DOUT);
    fill_i32_k<<<(Gn + 255) / 256, 256, 0, stream>>>(counts, 0, Gn);
    pool_accum_k<<<(Nn * DOUT + 255) / 256, 256, 0, stream>>>(node_emb, batch, gsum, gmaxb, counts);
    pool_concat_k<<<(Gn * 2 * DOUT + 255) / 256, 256, 0, stream>>>(gsum, gmaxb, counts, cc);

    // ---- graph MLP: graph_emb = relu(cc@Wg1+bg1)@Wg2+bg2
    float* graph_emb = (float*)d_out + (size_t)Nn * DOUT;
    gemm(cc, in(48), in(49), ghid, Gn, 2 * DOUT, DH, 1);
    gemm(ghid, in(50), in(51), graph_emb, Gn, DH, DOUT, 0);
}
